// NTXent_66245575574040
// MI455X (gfx1250) — matrix operand order repaired and checked by reading, not compiled, not run on hardware
//
#include <hip/hip_runtime.h>
#include <hip/hip_bf16.h>
#include <math.h>

// ---------------------------------------------------------------------------
// NT-Xent loss, fused: normalize -> f16 WMMA GEMM (LDS-staged B, async DMA,
// double-buffered) with masked exp-sum epilogue -> logsumexp finalize.
//
// Traffic plan: B panels staged once per 128-row workgroup via
// global_load_async_to_lds_b128 (ASYNCcnt) and shared by all 8 waves,
// cutting L2 reads from ~2 GB (no sharing) to ~256 MB.
//
// Numerics: logits are cos/tau <= 1/tau, so lse uses the fixed bound
// M = 1/tau (no online max); per-row partial exp-sums combine with float
// atomics.  The diagonal is always label-masked out and re-added analytically
// in the finalize kernel, keeping the WMMA loop branch-free.
// ---------------------------------------------------------------------------

typedef __attribute__((ext_vector_type(16))) _Float16 v16h;
typedef __attribute__((ext_vector_type(8)))  _Float16 v8h;
typedef __attribute__((ext_vector_type(8)))  float    v8f;

#define TAU      0.07f
#define INV_TAU  (1.0f / TAU)
#define K_EXP2   (INV_TAU * 1.4426950408889634f)   // log2(e)/tau
#define EPS_NRM  1e-6f
#define DDIM     256            // embedding dim
#define DDIMP    264            // padded LDS row (528 B = 132 DW == 4 mod 64 banks)
#define BM       128            // rows per workgroup (8 waves x 16-row tile)
#define CW       32             // columns staged per iteration (2 col tiles)
#define COLSPLIT 16             // column splits of the 8192-wide reduction

// ---------------------------------------------------------------------------
// Kernel 1: row-wise L2 normalize (torch clamp semantics) + cast to f16.
// ---------------------------------------------------------------------------
__global__ void k_normalize(const float* __restrict__ im,
                            const float* __restrict__ rec,
                            _Float16* __restrict__ imH,
                            _Float16* __restrict__ recH,
                            int n) {
    const float*  src = blockIdx.y ? rec  : im;
    _Float16*     dst = blockIdx.y ? recH : imH;
    const int lane = threadIdx.x & 31;
    const int row  = blockIdx.x * 8 + (threadIdx.x >> 5);
    if (row >= n) return;

    v8f x = *(const v8f*)(src + (size_t)row * DDIM + lane * 8);

    float ss = 0.0f;
#pragma unroll
    for (int i = 0; i < 8; ++i) ss += x[i] * x[i];
#pragma unroll
    for (int off = 16; off; off >>= 1) ss += __shfl_xor(ss, off, 32);

    const float scale = 1.0f / fmaxf(sqrtf(ss), EPS_NRM);

    v8h h;
#pragma unroll
    for (int i = 0; i < 8; ++i) h[i] = (_Float16)(x[i] * scale);
    *(v8h*)(dst + (size_t)row * DDIM + lane * 8) = h;
}

// ---------------------------------------------------------------------------
// Kernel 1b: diag[i] = (imH[i] . recH[i]) / tau   (one wave per row).
// ---------------------------------------------------------------------------
__global__ void k_diag(const _Float16* __restrict__ imH,
                       const _Float16* __restrict__ recH,
                       float* __restrict__ diag,
                       int n) {
    const int lane = threadIdx.x & 31;
    const int row  = blockIdx.x * 8 + (threadIdx.x >> 5);
    if (row >= n) return;

    v8h a = *(const v8h*)(imH  + (size_t)row * DDIM + lane * 8);
    v8h b = *(const v8h*)(recH + (size_t)row * DDIM + lane * 8);
    float d = 0.0f;
#pragma unroll
    for (int i = 0; i < 8; ++i) d += (float)a[i] * (float)b[i];
#pragma unroll
    for (int off = 16; off; off >>= 1) d += __shfl_xor(d, off, 32);
    if (lane == 0) diag[row] = d * INV_TAU;
}

// ---------------------------------------------------------------------------
// Kernel 2: fused GEMM + label-masked exp-sum, LDS-staged B.
// Block: 256 threads = 8 waves; wave w owns row tile w (16 rows).
// Grid:  (n/BM, COLSPLIT).  Per iteration: 32 columns staged to LDS by
// async DMA (double-buffered); each wave runs two interleaved 8-WMMA chains
// with B fragments software-pipelined one k-step ahead out of LDS.
// Epilogue works directly on the WMMA C register layout:
//   row = rowBase + v + 8*(lane>=16), col = colBase + (lane & 15).
// ---------------------------------------------------------------------------
__global__ void __launch_bounds__(256)
k_gemm_lse(const _Float16* __restrict__ imH,
           const _Float16* __restrict__ recH,
           const int* __restrict__ labels,
           float* __restrict__ rowsum,
           int n) {
    __shared__ _Float16 Bsh[2][CW * DDIMP];

    const int lane = threadIdx.x & 31;
    const int wave = threadIdx.x >> 5;     // 0..7 = row tile
    const int lo   = lane & 15;
    const int hi   = lane >> 4;            // 0/1

    const int rowBase  = blockIdx.x * BM + wave * 16;
    const int colStart = blockIdx.y * (n / COLSPLIT);
    const int nIter    = (n / COLSPLIT) / CW;   // 16

    // --- Preload A fragments: row = rowBase + lo, all 8 k-steps of 32.
    // A 16x32 layout (matching B): lane-half `hi` holds K = kb+16*hi..+15
    // -> one contiguous 16-half (32 B) chunk.
    const _Float16* aptr = imH + (size_t)(rowBase + lo) * DDIM;
    v16h afrag[DDIM / 32];
#pragma unroll
    for (int ks = 0; ks < DDIM / 32; ++ks) {
        const int kb = ks * 32;
        afrag[ks] = *(const v16h*)(aptr + kb + hi * 16);
    }

    // Labels for the 8 rows this lane's C registers touch.
    int rlab[8];
#pragma unroll
    for (int v = 0; v < 8; ++v) rlab[v] = labels[rowBase + v + 8 * hi];

    float acc[8];
#pragma unroll
    for (int v = 0; v < 8; ++v) acc[v] = 0.0f;

    // --- Async staging plan: wave w DMAs rec rows [w*4, w*4+4) of the CW=32
    // row panel; each lane moves 64 B as 4x b128 (offset applies to both the
    // global address and the LDS destination, per the async ISA pseudocode).
    const int srow   = wave * 4 + (lane >> 3);   // 0..31
    const int schunk = lane & 7;                 // 64 B chunks within a row
    const _Float16* gsrc0 =
        recH + (size_t)(colStart + srow) * DDIM + schunk * 32;
    const unsigned lds0 =
        (unsigned)(uintptr_t)&Bsh[0][srow * DDIMP + schunk * 32];
    const unsigned lds1 =
        (unsigned)(uintptr_t)&Bsh[1][srow * DDIMP + schunk * 32];

    auto stage = [&](int it) {
        const unsigned long long g =
            (unsigned long long)(uintptr_t)(gsrc0 + (size_t)it * CW * DDIM);
        const unsigned l = (it & 1) ? lds1 : lds0;
        asm volatile("global_load_async_to_lds_b128 %0, %1, off"
                     :: "v"(l), "v"(g) : "memory");
        asm volatile("global_load_async_to_lds_b128 %0, %1, off offset:16"
                     :: "v"(l), "v"(g) : "memory");
        asm volatile("global_load_async_to_lds_b128 %0, %1, off offset:32"
                     :: "v"(l), "v"(g) : "memory");
        asm volatile("global_load_async_to_lds_b128 %0, %1, off offset:48"
                     :: "v"(l), "v"(g) : "memory");
    };

    stage(0);
    asm volatile("s_wait_asynccnt 0x0" ::: "memory");
    __syncthreads();

    for (int it = 0; it < nIter; ++it) {
        if (it + 1 < nIter) stage(it + 1);     // DMA next panel during compute

        // Column labels for both tiles, loaded before the WMMA chains.
        const int gcol0 = colStart + it * CW + lo;
        const int clab0 = labels[gcol0];
        const int clab1 = labels[gcol0 + 16];

        // B 32x16 layout: lane = column, half `hi` holds K = kb+16*hi..+15.
        const _Float16* bp0 = &Bsh[it & 1][(lo)      * DDIMP + hi * 16];
        const _Float16* bp1 = &Bsh[it & 1][(16 + lo) * DDIMP + hi * 16];

        // Two interleaved accumulation chains, B software-pipelined +1 k-step.
        v8f c0 = {}, c1 = {};
        v16h b0 = *(const v16h*)(bp0);
        v16h b1 = *(const v16h*)(bp1);
#pragma unroll
        for (int ks = 0; ks < DDIM / 32; ++ks) {
            v16h n0, n1;
            if (ks + 1 < DDIM / 32) {
                n0 = *(const v16h*)(bp0 + (ks + 1) * 32);
                n1 = *(const v16h*)(bp1 + (ks + 1) * 32);
            }
            c0 = __builtin_amdgcn_wmma_f32_16x16x32_f16(
                     false, afrag[ks], false, b0, (short)0, c0, false, false);
            c1 = __builtin_amdgcn_wmma_f32_16x16x32_f16(
                     false, afrag[ks], false, b1, (short)0, c1, false, false);
            if (ks + 1 < DDIM / 32) { b0 = n0; b1 = n1; }
        }

        // Branch-free epilogue: keep iff labels differ (also excludes the
        // diagonal, re-added in k_finalize).
        // exp(x - 1/tau) = exp2((cos-1) * log2(e)/tau)  -- single mul + v_exp.
#pragma unroll
        for (int v = 0; v < 8; ++v) {
            const float e0 = exp2f((c0[v] - 1.0f) * K_EXP2);
            const float e1 = exp2f((c1[v] - 1.0f) * K_EXP2);
            acc[v] += (rlab[v] != clab0) ? e0 : 0.0f;
            acc[v] += (rlab[v] != clab1) ? e1 : 0.0f;
        }

        if (it + 1 < nIter) {
            asm volatile("s_wait_asynccnt 0x0" ::: "memory");
            __syncthreads();
        }
    }

    // Reduce each row's partial sum across the 16 lanes sharing that row,
    // then combine across workgroups with a float atomic.
#pragma unroll
    for (int v = 0; v < 8; ++v) {
        float a = acc[v];
#pragma unroll
        for (int off = 8; off; off >>= 1) a += __shfl_xor(a, off, 16);
        if (lo == 0) atomicAdd(&rowsum[rowBase + v + 8 * hi], a);
    }
}

// ---------------------------------------------------------------------------
// Kernel 3: loss = mean( 1/tau + log(rowsum_i + exp(diag_i - 1/tau)) - diag_i )
// ---------------------------------------------------------------------------
__global__ void k_finalize(const float* __restrict__ rowsum,
                           const float* __restrict__ diag,
                           float* __restrict__ out,
                           int n) {
    __shared__ float red[256];
    float local = 0.0f;
    for (int i = threadIdx.x; i < n; i += blockDim.x) {
        const float d = diag[i];
        const float s = rowsum[i] + __expf(d - INV_TAU);
        local += INV_TAU + logf(s) - d;
    }
    red[threadIdx.x] = local;
    __syncthreads();
    for (int s = 128; s; s >>= 1) {
        if ((int)threadIdx.x < s) red[threadIdx.x] += red[threadIdx.x + s];
        __syncthreads();
    }
    if (threadIdx.x == 0) out[0] = red[0] / (float)n;
}

// ---------------------------------------------------------------------------
extern "C" void kernel_launch(void* const* d_in, const int* in_sizes, int n_in,
                              void* d_out, int out_size, void* d_ws, size_t ws_size,
                              hipStream_t stream) {
    const float* im     = (const float*)d_in[0];
    const float* rec    = (const float*)d_in[1];
    const int*   labels = (const int*)d_in[2];
    float*       out    = (float*)d_out;

    const int n = in_sizes[2];          // 8192 rows
    // Workspace layout: imH | recH | rowsum | diag
    _Float16* imH    = (_Float16*)d_ws;
    _Float16* recH   = imH + (size_t)n * DDIM;
    float*    rowsum = (float*)(recH + (size_t)n * DDIM);
    float*    diagv  = rowsum + n;

    hipMemsetAsync(rowsum, 0, (size_t)n * sizeof(float), stream);

    dim3 g1(n / 8, 2);
    k_normalize<<<g1, 256, 0, stream>>>(im, rec, imH, recH, n);

    k_diag<<<n / 8, 256, 0, stream>>>(imH, recH, diagv, n);

    dim3 g2(n / BM, COLSPLIT);
    k_gemm_lse<<<g2, 256, 0, stream>>>(imH, recH, labels, rowsum, n);

    k_finalize<<<1, 256, 0, stream>>>(rowsum, diagv, out, n);
}
